// TemporalAttentionModule_90632399880607
// MI455X (gfx1250) — matrix operand order repaired and checked by reading, not compiled, not run on hardware
//
#include <hip/hip_runtime.h>
#include <hip/hip_bf16.h>

// ---- CDNA5 WMMA types ----
typedef __attribute__((ext_vector_type(16))) _Float16 v16h;
typedef __attribute__((ext_vector_type(8)))  _Float16 v8h;
typedef __attribute__((ext_vector_type(8)))  float    v8f;
typedef __attribute__((ext_vector_type(4)))  unsigned int u32x4;
typedef __attribute__((ext_vector_type(8)))  int      i32x8;
typedef __attribute__((ext_vector_type(4)))  int      i32x4;

static __device__ __forceinline__ v16h cat8(v8h lo, v8h hi) {
    return __builtin_shufflevector(lo, hi, 0,1,2,3,4,5,6,7,8,9,10,11,12,13,14,15);
}
static __device__ __forceinline__ v8f wmma_f16(v16h a, v16h b, v8f c) {
    // v_wmma_f32_16x16x32_f16  D = A*B + C   (wave32, fp32 accum)
    return __builtin_amdgcn_wmma_f32_16x16x32_f16(
        /*neg_a=*/false, a, /*neg_b=*/false, b,
        /*c_mod=*/(short)0, c, /*reuse_a=*/false, /*reuse_b=*/false);
}

// Problem constants (fixed by the reference).
constexpr int Cc = 2048;   // channels
constexpr int Tt = 256;    // timesteps
constexpr int Hh = 8;      // heads
constexpr int Dd = Cc / Hh; // 256 == Tt

// ---------------------------------------------------------------------
// TDM: 2D tile DMA  global -> LDS.  Tile rows of 512 B (64 x 8B elements),
// global row stride C*2 bytes = 512 elements, `rows` rows.
// D# built per CDNA5 ISA ch.8 (group0: count/lds/global/type, group1: dims).
// Issued once per wave (EXEC ignored by TDM); tracked by TENSORcnt.
// ---------------------------------------------------------------------
static __device__ __forceinline__ void tdm_load_2d(unsigned int ldsByteOff,
                                                   const void* gsrc, int rows) {
    const unsigned long long ga = (unsigned long long)(uintptr_t)gsrc;
    u32x4 g0;
    g0[0] = 1u;                                   // count=1 (valid), user mode
    g0[1] = ldsByteOff;                           // lds_addr (bytes)
    g0[2] = (unsigned int)ga;                     // global_addr[31:0]
    g0[3] = ((unsigned int)(ga >> 32) & 0x01FFFFFFu) | 0x80000000u; // [56:32] | type=2
    i32x8 g1;
    g1[0] = 0x00030000;            // wg_mask=0, data_size=3 (8B), no pad/iterate
    g1[1] = 64 << 16;              // tensor_dim0[15:0]=64 (at bits [79:64-32]... [31:16])
    g1[2] = rows << 16;            // tensor_dim0[31:16]=0 | tensor_dim1[15:0]=rows
    g1[3] = 64 << 16;              // tensor_dim1[31:16]=0 | tile_dim0=64
    g1[4] = rows;                  // tile_dim1=rows, tile_dim2=0 (2D)
    g1[5] = 512;                   // tensor_dim0_stride[31:0] = 512 elements (C*2 B)
    g1[6] = 0;                     // stride0[47:32]=0 | tensor_dim1_stride[15:0]=0
    g1[7] = 0;
    i32x4 gz = {0, 0, 0, 0};       // groups 2/3: dims/tiles zero -> unused (2D)
#if defined(__clang_major__) && (__clang_major__ >= 23)
    i32x8 gz8 = {0, 0, 0, 0, 0, 0, 0, 0};
    __builtin_amdgcn_tensor_load_to_lds(g0, g1, gz, gz, gz8, 0);
#else
    __builtin_amdgcn_tensor_load_to_lds(g0, g1, gz, gz, 0);
#endif
}

// =====================================================================
// Stage 1: projection  out(B,T,C) f16  = W(C,C) * x(B,C,T) + bias
// Block: 256 threads (8 waves). Tile: BM=128 (o) x BN=128 (t), BK=32 (c).
// Wave w computes a 16-row band (o) over all 128 cols (t): 8 WMMA frags.
// =====================================================================
constexpr int BM = 128, BN = 128, BK = 32;

__global__ __launch_bounds__(256)
void proj_kernel(const float* __restrict__ x, const float* __restrict__ W,
                 const float* __restrict__ bias, _Float16* __restrict__ out)
{
    __shared__ _Float16 ldsA[BM * BK];  // W tile, [m][k], k contiguous
    __shared__ _Float16 ldsB[BN * BK];  // x tile, transposed to [n][k]

    const int tid  = threadIdx.x;
    const int lane = tid & 31;
    const int wave = tid >> 5;
    const int lane16   = lane & 15;
    const int laneHalf = lane >> 4;          // 0: lanes 0-15, 1: lanes 16-31

    const int mTile = blockIdx.x >> 1;       // 16 M-tiles
    const int nTile = blockIdx.x & 1;        // 2  N-tiles
    const int b     = blockIdx.y;
    const int oBase = mTile * BM;
    const int tBase = nTile * BN;

    const float* xb = x + (size_t)b * Cc * Tt;

    v8f zero = {};
    v8f acc[8];
#pragma unroll
    for (int f = 0; f < 8; ++f) acc[f] = zero;

    for (int kk = 0; kk < Cc; kk += BK) {
        // ---- load A: W[oBase..+127, kk..+31] -> ldsA[m][k] (f16) ----
        {
            const int row = tid >> 1;                 // 0..127
            const int seg = (tid & 1) << 4;           // 0 or 16
            const float* src = W + (size_t)(oBase + row) * Cc + kk + seg;
            __builtin_prefetch(src + BK, 0, 0);       // next K tile -> global_prefetch
            float4 a0 = ((const float4*)src)[0];
            float4 a1 = ((const float4*)src)[1];
            float4 a2 = ((const float4*)src)[2];
            float4 a3 = ((const float4*)src)[3];
            v8h h0, h1;
            h0[0]=(_Float16)a0.x; h0[1]=(_Float16)a0.y; h0[2]=(_Float16)a0.z; h0[3]=(_Float16)a0.w;
            h0[4]=(_Float16)a1.x; h0[5]=(_Float16)a1.y; h0[6]=(_Float16)a1.z; h0[7]=(_Float16)a1.w;
            h1[0]=(_Float16)a2.x; h1[1]=(_Float16)a2.y; h1[2]=(_Float16)a2.z; h1[3]=(_Float16)a2.w;
            h1[4]=(_Float16)a3.x; h1[5]=(_Float16)a3.y; h1[6]=(_Float16)a3.z; h1[7]=(_Float16)a3.w;
            _Float16* dst = &ldsA[row * BK + seg];
            *(v8h*)dst       = h0;
            *(v8h*)(dst + 8) = h1;
        }
        // ---- load B: x[kk..+31, tBase..+127] -> ldsB[n][k] (transposed) ----
        {
            const int crow = tid >> 3;                // 0..31  (k row)
            const int tseg = (tid & 7) << 4;          // 0..112 (t offset)
            const float* src = xb + (size_t)(kk + crow) * Tt + tBase + tseg;
            float4 b0 = ((const float4*)src)[0];
            float4 b1 = ((const float4*)src)[1];
            float4 b2 = ((const float4*)src)[2];
            float4 b3 = ((const float4*)src)[3];
            float vals[16] = { b0.x,b0.y,b0.z,b0.w, b1.x,b1.y,b1.z,b1.w,
                               b2.x,b2.y,b2.z,b2.w, b3.x,b3.y,b3.z,b3.w };
#pragma unroll
            for (int j = 0; j < 16; ++j)
                ldsB[(tseg + j) * BK + crow] = (_Float16)vals[j];
        }
        __syncthreads();

        // ---- A fragment: 16x32, rows = wave band; per-lane K layout per ISA ----
        const _Float16* ap = &ldsA[(wave * 16 + lane16) * BK + laneHalf * 8];
        v16h afrag = cat8(*(const v8h*)ap, *(const v8h*)(ap + 16));
#pragma unroll
        for (int f = 0; f < 8; ++f) {
            const _Float16* bp = &ldsB[(f * 16 + lane16) * BK + laneHalf * 8];
            v16h bfrag = cat8(*(const v8h*)bp, *(const v8h*)(bp + 16));
            acc[f] = wmma_f16(afrag, bfrag, acc[f]);
        }
        __syncthreads();
    }

    // ---- epilogue: add bias, convert, store f16 to out(B,T,C) ----
    const float* bp = bias + oBase + wave * 16 + laneHalf * 8;
    float bvv[8];
#pragma unroll
    for (int r = 0; r < 8; ++r) bvv[r] = bp[r];

    _Float16* ob = out + (size_t)b * Tt * Cc;
#pragma unroll
    for (int f = 0; f < 8; ++f) {
        const int t = tBase + f * 16 + lane16;
        v8h h;
#pragma unroll
        for (int r = 0; r < 8; ++r) h[r] = (_Float16)(acc[f][r] + bvv[r]);
        // 8 consecutive o's -> one 16B store
        *(v8h*)(ob + (size_t)t * Cc + oBase + wave * 16 + laneHalf * 8) = h;
    }
}

// =====================================================================
// Stage 2: attention for one (b, h, i-half). 320 KB dynamic LDS:
//   [0..64K)    ldsQ  [128][256] f16   (later reused as attn P)
//   [64..192K)  ldsK  [256][256] f16   layout [j][d]
//   [192..320K) ldsV  [256][256] f16   layout [j][k]
// Tiles staged by the Tensor Data Mover (waves 0/1/2 issue Q/K/V DMAs).
// Each wave owns a 16-row (i) band; full 256-wide rows resident -> plain softmax.
// =====================================================================
__global__ __launch_bounds__(256)
void attn_kernel(const _Float16* __restrict__ q, const _Float16* __restrict__ k,
                 const _Float16* __restrict__ v, float* __restrict__ out)
{
    extern __shared__ char smem[];
    _Float16* ldsQ = (_Float16*)(smem);            // LDS byte offset 0
    _Float16* ldsK = (_Float16*)(smem + 65536);
    _Float16* ldsV = (_Float16*)(smem + 196608);

    const int tid  = threadIdx.x;
    const int lane = tid & 31;
    const int wave = tid >> 5;
    const int lane16   = lane & 15;
    const int laneHalf = lane >> 4;

    const int ih = blockIdx.x;          // 0/1: which 128-row i-chunk
    const int h  = blockIdx.y;
    const int b  = blockIdx.z;
    const int i0 = ih * 128;

    // ws layout is (B,T,C): row stride C halves, head offset h*256, d contiguous.
    const size_t baseQ  = ((size_t)b * Tt + i0) * Cc + (size_t)h * Dd;
    const size_t baseKV = ((size_t)b * Tt) * Cc + (size_t)h * Dd;

    // ---- stage tiles via Tensor Data Mover (one DMA per wave 0/1/2) ----
    if (wave == 0)      tdm_load_2d(0u,      q + baseQ,  128);
    else if (wave == 1) tdm_load_2d(65536u,  k + baseKV, 256);
    else if (wave == 2) tdm_load_2d(196608u, v + baseKV, 256);
    if (wave < 3) __builtin_amdgcn_s_wait_tensorcnt(0);
    __syncthreads();

    v8f zero = {};
    v8f sAcc[16];
#pragma unroll
    for (int f = 0; f < 16; ++f) sAcc[f] = zero;

    // ---- S = Q . K^T over d (K-dim 256, step 32) ----
    for (int d0 = 0; d0 < 256; d0 += 32) {
        const _Float16* ap = ldsQ + (wave * 16 + lane16) * 256 + d0 + laneHalf * 8;
        v16h afrag = cat8(*(const v8h*)ap, *(const v8h*)(ap + 16));
#pragma unroll
        for (int f = 0; f < 16; ++f) {
            const _Float16* bp = ldsK + (f * 16 + lane16) * 256 + d0 + laneHalf * 8;
            v16h bfrag = cat8(*(const v8h*)bp, *(const v8h*)(bp + 16));
            sAcc[f] = wmma_f16(afrag, bfrag, sAcc[f]);
        }
    }
    __syncthreads();   // everyone done reading ldsQ before we overwrite with P

    // ---- softmax over j (full 256-wide rows).  scale = D^-0.5 = 1/16 ----
    // C/D frag: lanes 0-15 hold row (vgpr r), lanes 16-31 hold row (8+r);
    // row elements live across the 16 lanes of each half -> xor-reduce 1,2,4,8.
    const float scale = 0.0625f;
    float rmax[8], rsum[8];
#pragma unroll
    for (int r = 0; r < 8; ++r) rmax[r] = -3.4e38f;
#pragma unroll
    for (int f = 0; f < 16; ++f)
#pragma unroll
        for (int r = 0; r < 8; ++r) rmax[r] = fmaxf(rmax[r], sAcc[f][r]);
#pragma unroll
    for (int r = 0; r < 8; ++r) {
        rmax[r] = fmaxf(rmax[r], __shfl_xor(rmax[r], 1, 32));
        rmax[r] = fmaxf(rmax[r], __shfl_xor(rmax[r], 2, 32));
        rmax[r] = fmaxf(rmax[r], __shfl_xor(rmax[r], 4, 32));
        rmax[r] = fmaxf(rmax[r], __shfl_xor(rmax[r], 8, 32));
        rsum[r] = 0.0f;
    }
#pragma unroll
    for (int f = 0; f < 16; ++f)
#pragma unroll
        for (int r = 0; r < 8; ++r) {
            float p = __expf((sAcc[f][r] - rmax[r]) * scale);
            sAcc[f][r] = p;
            rsum[r] += p;
        }
#pragma unroll
    for (int r = 0; r < 8; ++r) {
        rsum[r] += __shfl_xor(rsum[r], 1, 32);
        rsum[r] += __shfl_xor(rsum[r], 2, 32);
        rsum[r] += __shfl_xor(rsum[r], 4, 32);
        rsum[r] += __shfl_xor(rsum[r], 8, 32);
        rsum[r] = 1.0f / rsum[r];
    }
    // write P (f16) into ldsQ region as A-layout [i][k]
#pragma unroll
    for (int f = 0; f < 16; ++f)
#pragma unroll
        for (int r = 0; r < 8; ++r) {
            const int row = wave * 16 + laneHalf * 8 + r;
            ldsQ[row * 256 + f * 16 + lane16] = (_Float16)(sAcc[f][r] * rsum[r]);
        }
    __syncthreads();

    // ---- out = P . V  over k (K-dim 256, step 32); reuse sAcc as accumulators ----
#pragma unroll
    for (int f = 0; f < 16; ++f) sAcc[f] = zero;
    for (int kk = 0; kk < 256; kk += 32) {
        const _Float16* ap = ldsQ + (wave * 16 + lane16) * 256 + kk + laneHalf * 8;
        v16h afrag = cat8(*(const v8h*)ap, *(const v8h*)(ap + 16));
#pragma unroll
        for (int f = 0; f < 16; ++f) {
            const _Float16* bp = ldsV + (f * 16 + lane16) * 256 + kk + laneHalf * 8;
            v16h bfrag = cat8(*(const v8h*)bp, *(const v8h*)(bp + 16));
            sAcc[f] = wmma_f16(afrag, bfrag, sAcc[f]);
        }
    }

    // ---- store fp32 to out(B,C,T): idx = ((b*C + h*T + i) * T) + j ----
    float* ob = out + ((size_t)b * Cc + (size_t)h * Tt + i0 + wave * 16) * Tt;
#pragma unroll
    for (int f = 0; f < 16; ++f)
#pragma unroll
        for (int r = 0; r < 8; ++r)
            ob[(size_t)(laneHalf * 8 + r) * Tt + f * 16 + lane16] = sAcc[f][r];
}

// =====================================================================
extern "C" void kernel_launch(void* const* d_in, const int* in_sizes, int n_in,
                              void* d_out, int out_size, void* d_ws, size_t ws_size,
                              hipStream_t stream)
{
    const float* x  = (const float*)d_in[0];
    const float* Wq = (const float*)d_in[1];
    const float* bq = (const float*)d_in[2];
    const float* Wk = (const float*)d_in[3];
    const float* bk = (const float*)d_in[4];
    const float* Wv = (const float*)d_in[5];
    const float* bv = (const float*)d_in[6];
    // heads fixed at 8 per the reference module's shapes.

    const size_t perProj = (size_t)64 * Tt * Cc;       // B*T*C halves
    _Float16* wsq = (_Float16*)d_ws;
    _Float16* wsk = wsq + perProj;
    _Float16* wsv = wsk + perProj;

    dim3 blk(256);
    dim3 g1(32, 64);            // 16 M-tiles * 2 N-tiles, B=64
    proj_kernel<<<g1, blk, 0, stream>>>(x, Wq, bq, wsq);
    proj_kernel<<<g1, blk, 0, stream>>>(x, Wk, bk, wsk);
    proj_kernel<<<g1, blk, 0, stream>>>(x, Wv, bv, wsv);

    dim3 g2(2, Hh, 64);         // i-half, head, batch
    attn_kernel<<<g2, blk, 327680, stream>>>(wsq, wsk, wsv, (float*)d_out);
}
